// LocalSpatialAttentionPatch_58119497450443
// MI455X (gfx1250) — matrix operand order repaired and checked by reading, not compiled, not run on hardware
//
#include <hip/hip_runtime.h>

typedef __attribute__((ext_vector_type(16))) _Float16 v16h;
typedef __attribute__((ext_vector_type(8)))  _Float16 v8h;
typedef __attribute__((ext_vector_type(8)))  float    v8f;

#define HEADS 4
#define HD    32
#define CDIM  128           // input channels C
#define CK    128           // HEADS*HD
#define IMG_H 96
#define IMG_W 96
#define BATCH 4
#define NPIX  (BATCH*IMG_H*IMG_W)   // 36864
#define NS    25
#define PW    5

// ---------------------------------------------------------------------------
// Prep: convert the four 128x128 f32 weight matrices to f16, TRANSPOSED
// (WT[n][k]), so WMMA B-fragments are contiguous 16-half (32B) loads.
// ---------------------------------------------------------------------------
__global__ void __launch_bounds__(256)
prep_weights(const float* __restrict__ Wq, const float* __restrict__ Wk,
             const float* __restrict__ Wv, const float* __restrict__ Wo,
             _Float16* __restrict__ WqT, _Float16* __restrict__ WkT,
             _Float16* __restrict__ WvT, _Float16* __restrict__ WoT)
{
    int t = blockIdx.x * blockDim.x + threadIdx.x;
    for (int i = t; i < CDIM * CK; i += blockDim.x * gridDim.x) {
        int k = i / CK, n = i % CK;          // source: row k (fan_in), col n
        WqT[n * CDIM + k] = (_Float16)Wq[i];
        WkT[n * CDIM + k] = (_Float16)Wk[i];
        WvT[n * CDIM + k] = (_Float16)Wv[i];
        WoT[n * CDIM + k] = (_Float16)Wo[i];
    }
}

// ---------------------------------------------------------------------------
// QKV projection GEMM:  [NPIX x 128] (f32) @ [128 x 128] -> f16 maps.
// grid = (NPIX/32, 3 projections), block = 256 (8 waves).
// Block covers 32 rows x 128 cols. Wave w: row-tile = w>>2, N-pair = w&3;
// one A fragment feeds TWO 16x16x32 WMMAs (adjacent N tiles) per K step.
// Q projection (p==0) adds pos_emb[qidx(y,x)] in the f32 epilogue; since
// 16 | 96, a 16-row tile never crosses an image row -> yy/xx_base are
// wave-uniform (SALU), no per-lane division.
// ---------------------------------------------------------------------------
__global__ void __launch_bounds__(256)
gemm_qkv(const float* __restrict__ x,
         const _Float16* __restrict__ WqT, const _Float16* __restrict__ WkT,
         const _Float16* __restrict__ WvT,
         const float* __restrict__ pos_emb,
         _Float16* __restrict__ Qm, _Float16* __restrict__ Km,
         _Float16* __restrict__ Vm)
{
    const int p = blockIdx.y;               // 0=Q, 1=K, 2=V
    const _Float16* __restrict__ WT = (p == 0) ? WqT : (p == 1) ? WkT : WvT;
    _Float16* __restrict__ dst      = (p == 0) ? Qm  : (p == 1) ? Km  : Vm;

    const int wave  = threadIdx.x >> 5;
    const int lane  = threadIdx.x & 31;
    const int r     = lane & 15;            // A-row / B-col index within tile
    const int g     = lane >> 4;            // lane half
    const int rt    = wave >> 2;            // row tile within block (0..1)
    const int npair = wave & 3;             // N-pair (0..3) -> cols npair*32

    const int R0    = blockIdx.x * 32 + rt * 16;   // wave-uniform row base
    const int row   = R0 + r;
    const int ncol0 = npair * 32;
    const int ncol1 = ncol0 + 16;

    const float*    __restrict__ arow  = x  + (size_t)row * CDIM;
    const _Float16* __restrict__ brow0 = WT + (size_t)(ncol0 + r) * CDIM;
    const _Float16* __restrict__ brow1 = WT + (size_t)(ncol1 + r) * CDIM;

    v8f acc0 = {}, acc1 = {};
#pragma unroll
    for (int ks = 0; ks < 4; ++ks) {
        const int kb = ks * 32;
        // A fragment: lane holds 16 consecutive K values starting at g*16
        // (convert f32 -> f16 on the fly).
        v16h a;
        const float* a0 = arow + kb + g * 16;
        const float* a1 = arow + kb + g * 16 + 8;
#pragma unroll
        for (int j = 0; j < 8; ++j) {
            a[j]     = (_Float16)a0[j];
            a[8 + j] = (_Float16)a1[j];
        }
        // B fragments: lane holds column (ncolX+r), 16 consecutive K values
        // starting at kb + g*16 (contiguous 32B from transposed weights).
        v8h b0l = *(const v8h*)(brow0 + kb + g * 16);
        v8h b0h = *(const v8h*)(brow0 + kb + g * 16 + 8);
        v8h b1l = *(const v8h*)(brow1 + kb + g * 16);
        v8h b1h = *(const v8h*)(brow1 + kb + g * 16 + 8);
        v16h b0, b1;
#pragma unroll
        for (int j = 0; j < 8; ++j) {
            b0[j] = b0l[j]; b0[8 + j] = b0h[j];
            b1[j] = b1l[j]; b1[8 + j] = b1h[j];
        }
        acc0 = __builtin_amdgcn_wmma_f32_16x16x32_f16(
                   false, a, false, b0, (short)0, acc0, false, false);
        acc1 = __builtin_amdgcn_wmma_f32_16x16x32_f16(
                   false, a, false, b1, (short)0, acc1, false, false);
    }

    // Epilogue: D layout -> M = v + 8*g, N = lane&15.
    const int col0 = ncol0 + r;
    const int col1 = ncol1 + r;
    // Wave-uniform image coordinates (16-row tile stays inside one row).
    const int yy      = (R0 / IMG_W) % IMG_H;
    const int xx_base = R0 % IMG_W;
    const int cy      = min(max(yy, 2), IMG_H - 3);
#pragma unroll
    for (int v = 0; v < 8; ++v) {
        const int mrow = 8 * g + v;
        const int drow = R0 + mrow;
        float val0 = acc0[v];
        float val1 = acc1[v];
        if (p == 0) {
            const int xx  = xx_base + mrow;
            const int cx  = min(max(xx, 2), IMG_W - 3);
            const int qid = (yy - cy + 2) * PW + (xx - cx + 2);
            val0 += pos_emb[qid * CK + col0];
            val1 += pos_emb[qid * CK + col1];
        }
        dst[(size_t)drow * CK + col0] = (_Float16)val0;
        dst[(size_t)drow * CK + col1] = (_Float16)val1;
    }
}

// ---------------------------------------------------------------------------
// Attention core: one thread per (pixel, head). 25-neighbor gather from the
// dense K/V maps (everything resident in the 192MB L2), register softmax.
// K_eff[n] = Kmap[sy,sx] + pos_emb[n]  (per the reference).
// All f16 gathers are explicit 16B vector loads -> global_load_b128.
// ---------------------------------------------------------------------------
__global__ void __launch_bounds__(256)
attn(const _Float16* __restrict__ Qm, const _Float16* __restrict__ Km,
     const _Float16* __restrict__ Vm, const float* __restrict__ pos_emb,
     _Float16* __restrict__ att)
{
    const int gid = blockIdx.x * blockDim.x + threadIdx.x;
    if (gid >= NPIX * HEADS) return;
    const int h   = gid & (HEADS - 1);
    const int pix = gid >> 2;
    const int xx  = pix % IMG_W;
    const int yy  = (pix / IMG_W) % IMG_H;
    const int bb  = pix / (IMG_W * IMG_H);

    float q[HD];
    {
        const v8h* qv = (const v8h*)(Qm + (size_t)pix * CK + h * HD);
#pragma unroll
        for (int c = 0; c < HD / 8; ++c) {
            v8h t = qv[c];
#pragma unroll
            for (int j = 0; j < 8; ++j) q[c * 8 + j] = (float)t[j];
        }
    }

    const int cy = min(max(yy, 2), IMG_H - 3);
    const int cx = min(max(xx, 2), IMG_W - 3);
    const float scale = 0.17677669529663687f;   // 1/sqrt(32)

    float sc[NS];
    float m = -1e30f;
#pragma unroll
    for (int dy = 0; dy < PW; ++dy) {
#pragma unroll
        for (int dx = 0; dx < PW; ++dx) {
            const int n   = dy * PW + dx;
            const int kpx = (bb * IMG_H + (cy + dy - 2)) * IMG_W + (cx + dx - 2);
            const v8h*   kv = (const v8h*)(Km + (size_t)kpx * CK + h * HD);
            const float* pe = pos_emb + n * CK + h * HD;
            float s = 0.f;
#pragma unroll
            for (int c = 0; c < HD / 8; ++c) {
                v8h t = kv[c];
#pragma unroll
                for (int j = 0; j < 8; ++j)
                    s += q[c * 8 + j] * ((float)t[j] + pe[c * 8 + j]);
            }
            s *= scale;
            sc[n] = s;
            m = fmaxf(m, s);
        }
    }

    float sum = 0.f;
#pragma unroll
    for (int n = 0; n < NS; ++n) { float e = __expf(sc[n] - m); sc[n] = e; sum += e; }
    const float inv = 1.f / sum;

    float o[HD];
#pragma unroll
    for (int d = 0; d < HD; ++d) o[d] = 0.f;
#pragma unroll
    for (int dy = 0; dy < PW; ++dy) {
#pragma unroll
        for (int dx = 0; dx < PW; ++dx) {
            const int n   = dy * PW + dx;
            const int kpx = (bb * IMG_H + (cy + dy - 2)) * IMG_W + (cx + dx - 2);
            const v8h* vv = (const v8h*)(Vm + (size_t)kpx * CK + h * HD);
            const float wn = sc[n] * inv;
#pragma unroll
            for (int c = 0; c < HD / 8; ++c) {
                v8h t = vv[c];
#pragma unroll
                for (int j = 0; j < 8; ++j) o[c * 8 + j] += wn * (float)t[j];
            }
        }
    }

    _Float16* op = att + (size_t)pix * CK + h * HD;
#pragma unroll
    for (int d = 0; d < HD; ++d) op[d] = (_Float16)o[d];
}

// ---------------------------------------------------------------------------
// Output projection GEMM: att(f16) @ WoutT -> d_out (f32).
// Same two-N-tiles-per-wave structure as gemm_qkv.
// ---------------------------------------------------------------------------
__global__ void __launch_bounds__(256)
gemm_out(const _Float16* __restrict__ att, const _Float16* __restrict__ WoT,
         float* __restrict__ out)
{
    const int wave  = threadIdx.x >> 5;
    const int lane  = threadIdx.x & 31;
    const int r     = lane & 15;
    const int g     = lane >> 4;
    const int rt    = wave >> 2;
    const int npair = wave & 3;

    const int R0    = blockIdx.x * 32 + rt * 16;
    const int row   = R0 + r;
    const int ncol0 = npair * 32;
    const int ncol1 = ncol0 + 16;

    const _Float16* __restrict__ arow  = att + (size_t)row * CK;
    const _Float16* __restrict__ brow0 = WoT + (size_t)(ncol0 + r) * CDIM;
    const _Float16* __restrict__ brow1 = WoT + (size_t)(ncol1 + r) * CDIM;

    v8f acc0 = {}, acc1 = {};
#pragma unroll
    for (int ks = 0; ks < 4; ++ks) {
        const int kb = ks * 32;
        v8h a0 = *(const v8h*)(arow + kb + g * 16);
        v8h a1 = *(const v8h*)(arow + kb + g * 16 + 8);
        v8h b0l = *(const v8h*)(brow0 + kb + g * 16);
        v8h b0h = *(const v8h*)(brow0 + kb + g * 16 + 8);
        v8h b1l = *(const v8h*)(brow1 + kb + g * 16);
        v8h b1h = *(const v8h*)(brow1 + kb + g * 16 + 8);
        v16h a, b0, b1;
#pragma unroll
        for (int j = 0; j < 8; ++j) {
            a[j]  = a0[j];  a[8 + j]  = a1[j];
            b0[j] = b0l[j]; b0[8 + j] = b0h[j];
            b1[j] = b1l[j]; b1[8 + j] = b1h[j];
        }
        acc0 = __builtin_amdgcn_wmma_f32_16x16x32_f16(
                   false, a, false, b0, (short)0, acc0, false, false);
        acc1 = __builtin_amdgcn_wmma_f32_16x16x32_f16(
                   false, a, false, b1, (short)0, acc1, false, false);
    }

    const int col0 = ncol0 + r;
    const int col1 = ncol1 + r;
#pragma unroll
    for (int v = 0; v < 8; ++v) {
        const int drow = R0 + 8 * g + v;
        out[(size_t)drow * CDIM + col0] = acc0[v];
        out[(size_t)drow * CDIM + col1] = acc1[v];
    }
}

// ---------------------------------------------------------------------------
extern "C" void kernel_launch(void* const* d_in, const int* in_sizes, int n_in,
                              void* d_out, int out_size, void* d_ws, size_t ws_size,
                              hipStream_t stream)
{
    (void)in_sizes; (void)n_in; (void)out_size; (void)ws_size;
    const float* x    = (const float*)d_in[0];
    const float* Wq   = (const float*)d_in[1];
    const float* Wk   = (const float*)d_in[2];
    const float* Wv   = (const float*)d_in[3];
    const float* Wo   = (const float*)d_in[4];
    const float* pos  = (const float*)d_in[5];

    char* ws = (char*)d_ws;
    const size_t WBYTES = (size_t)CDIM * CK * sizeof(_Float16);   // 32 KB
    const size_t MBYTES = (size_t)NPIX * CK * sizeof(_Float16);   // 9.4 MB
    _Float16* WqT = (_Float16*)(ws);
    _Float16* WkT = (_Float16*)(ws + 1 * WBYTES);
    _Float16* WvT = (_Float16*)(ws + 2 * WBYTES);
    _Float16* WoT = (_Float16*)(ws + 3 * WBYTES);
    _Float16* Qm  = (_Float16*)(ws + 4 * WBYTES);
    _Float16* Km  = (_Float16*)(ws + 4 * WBYTES + 1 * MBYTES);
    _Float16* Vm  = (_Float16*)(ws + 4 * WBYTES + 2 * MBYTES);
    _Float16* at  = (_Float16*)(ws + 4 * WBYTES + 3 * MBYTES);

    prep_weights<<<64, 256, 0, stream>>>(Wq, Wk, Wv, Wo, WqT, WkT, WvT, WoT);

    dim3 g1(NPIX / 32, 3);
    gemm_qkv<<<g1, 256, 0, stream>>>(x, WqT, WkT, WvT, pos, Qm, Km, Vm);

    attn<<<(NPIX * HEADS) / 256, 256, 0, stream>>>(Qm, Km, Vm, pos, at);

    gemm_out<<<NPIX / 32, 256, 0, stream>>>(at, WoT, (float*)d_out);
}
